// LSTM_52767968198994
// MI455X (gfx1250) — matrix operand order repaired and checked by reading, not compiled, not run on hardware
//
#include <hip/hip_runtime.h>

// Problem constants (from reference)
#define HH   256          // hidden
#define DD   4096         // input feature dim
#define BB   64           // batch
#define TT   256          // timesteps
#define GG   1024         // 4*H gate width
#define MM   (BB*TT)      // 16384 rows of the input GEMM
#define NWG  4            // workgroups cooperating on the recurrence

typedef __attribute__((ext_vector_type(16))) _Float16 v16h;
typedef __attribute__((ext_vector_type(8)))  _Float16 v8h;
typedef __attribute__((ext_vector_type(4)))  _Float16 v4h;
typedef __attribute__((ext_vector_type(8)))  float    v8f;
typedef __attribute__((ext_vector_type(4)))  float    v4f;

static __device__ __forceinline__ v16h cat16(v8h lo, v8h hi) {
  return __builtin_shufflevector(lo, hi, 0,1,2,3,4,5,6,7,8,9,10,11,12,13,14,15);
}

static __device__ __forceinline__ float sigm(float v) {
  return 1.0f / (1.0f + __expf(-v));
}

// -------------------------------------------------------------------------
// Phase 2: 4 persistent WGs, 512 thr (16 waves) each.
// WG w owns gate cols {g*256 + w*64 + 0..63}.  Each wave owns ONE 16-col
// n-tile -> its w_hh B-fragments = 8 v16h = 64 VGPRs, resident all 256 steps
// (verified spill-free in round 4 disasm).
// Per step: GEMM (512 wmma/WG) -> local barrier -> activation -> global
// generation barrier (atomic + s_sleep spin) -> acquire fence by ALL threads
// (CU-mode: both CUs' L0 must be invalidated) -> copy h (32KB, L2) into LDS.
// h is parity double-buffered in global memory to make the exchange race-free.
// -------------------------------------------------------------------------
#define GLS 272   // gates slice row stride (halfs): 256 cols + pad
#define HST 264   // h row stride in LDS (halfs)
#define CLS 68    // c slice row stride (f32)
#define SM2 (64*GLS*2 + 64*HST*2 + 64*CLS*4)   // 34816+33792+17408 = 86016

__global__ __launch_bounds__(512)
void k_lstm(const float* __restrict__ pre, const _Float16* __restrict__ wb,
            const float* __restrict__ w_fc, const float* __restrict__ b_fc,
            _Float16* __restrict__ hg,     // [2][64][256] f16 double buffer
            float* __restrict__ hfin,      // [64][256] f32 final h
            int* __restrict__ bar,         // generation counter (zeroed per launch)
            float* __restrict__ out)
{
  extern __shared__ char smem[];
  _Float16* gl = (_Float16*)smem;                          // gates slice
  _Float16* hs = (_Float16*)(smem + 64 * GLS * 2);         // full h (f16)
  float*    cs = (float*)   (smem + 64 * GLS * 2 + 64 * HST * 2);  // c slice

  const int w     = blockIdx.x;       // 0..3
  const int tid   = threadIdx.x;      // 0..511
  const int lane  = tid & 31;
  const int wave  = tid >> 5;         // 0..15
  const int row16 = lane & 15;
  const int half  = lane >> 4;
  const int q     = wave & 3;         // tile-in-chunk
  const int g4    = wave >> 2;        // gate group 0..3
  const int ntg   = g4 * 16 + w * 4 + q;   // global n-tile (0..63)

  // init: h0 = 0, c0 = 0
  for (int i = tid; i < 64 * HST; i += 512) hs[i] = (_Float16)0.0f;
  for (int i = tid; i < 64 * CLS; i += 512) cs[i] = 0.0f;

  // hoist this wave's w_hh n-tile into registers (8 frags, 64 VGPRs)
  v16h Bf[8];
  #pragma unroll
  for (int ki = 0; ki < 8; ++ki) {
    const _Float16* p = wb + (((size_t)ntg * 8 + ki) * 32 + lane) * 16;
    Bf[ki] = cat16(*(const v8h*)p, *(const v8h*)(p + 8));
  }

  // activation mapping: thread owns slice col jl, rows m = mgp*8 .. +7
  const int jl  = tid & 63;
  const int mgp = tid >> 6;           // 0..7

  __syncthreads();

  for (int t = 0; t < TT; ++t) {
    // ---- GEMM: gates_slice = h @ w_hh[slice]^T  (M=64,N=256,K=256) ----
    const int lcol = wave * 16 + row16;          // local gates column
    #pragma unroll
    for (int tm = 0; tm < 4; ++tm) {
      v8f acc = {};
      #pragma unroll
      for (int ki = 0; ki < 8; ++ki) {
        const _Float16* p = hs + (tm * 16 + row16) * HST + ki * 32 + half * 16;
        v16h a = cat16(*(const v8h*)p, *(const v8h*)(p + 8));
        acc = __builtin_amdgcn_wmma_f32_16x16x32_f16(
            false, a, false, Bf[ki], (short)0, acc, false, false);
      }
      const int mb = tm * 16 + half * 8;
      #pragma unroll
      for (int e = 0; e < 8; ++e)
        gl[(mb + e) * GLS + lcol] = (_Float16)acc[e];
    }
    __syncthreads();   // gates slice ready; all hs reads done

    // ---- elementwise LSTM cell for this WG's 64 j-columns ----
    // local col layout: gate g of column jl sits at  g*64 + jl
    _Float16* hgw = hg + (size_t)((t + 1) & 1) * (64 * 256);
    #pragma unroll 4
    for (int r = 0; r < 8; ++r) {
      const int m = mgp * 8 + r;
      const float* pm = pre + ((size_t)m * TT + t) * GG + w * 64;
      float gi = (float)gl[m * GLS + jl      ] + pm[jl      ];
      float gf = (float)gl[m * GLS + jl +  64] + pm[jl + 256];
      float gc = (float)gl[m * GLS + jl + 128] + pm[jl + 512];
      float go = (float)gl[m * GLS + jl + 192] + pm[jl + 768];
      float iv = sigm(gi), fv = sigm(gf), gv = tanhf(gc), ov = sigm(go);
      float cn = fv * cs[m * CLS + jl] + iv * gv;
      cs[m * CLS + jl] = cn;
      float hv = ov * tanhf(cn);
      hgw[m * 256 + w * 64 + jl] = (_Float16)hv;
      if (t == TT - 1) hfin[m * 256 + w * 64 + jl] = hv;
    }

    // ---- device-wide generation barrier across the 4 WGs ----
    __threadfence();            // release: publish h stores (all threads)
    __syncthreads();
    if (tid == 0) {
      __hip_atomic_fetch_add(bar, 1, __ATOMIC_RELEASE, __HIP_MEMORY_SCOPE_AGENT);
      while (__hip_atomic_load(bar, __ATOMIC_ACQUIRE, __HIP_MEMORY_SCOPE_AGENT)
             < NWG * (t + 1))
        __builtin_amdgcn_s_sleep(2);
    }
    __syncthreads();
    __threadfence();            // acquire for every wave (covers both CUs' L0)

    // ---- copy h(t+1) [64][256] f16 (32KB, L2) into LDS ----
    #pragma unroll
    for (int i2 = 0; i2 < 4; ++i2) {
      int cidx = i2 * 512 + tid;            // 2048 x 16B chunks
      int m = cidx >> 3, off = (cidx & 7) * 8;
      *(v8h*)(hs + m * HST + off) = *(const v8h*)(hgw + m * 256 + off);
    }
    __syncthreads();
  }

  // ---- final FC (WG 0): out[b] = h_last[b] . w_fc + b_fc ----
  if (w == 0 && tid < BB) {
    float s = b_fc[0];
    #pragma unroll 4
    for (int jj = 0; jj < HH; ++jj)
      s += hfin[tid * 256 + jj] * w_fc[jj];
    out[tid] = s;
  }
}

// -------------------------------------------------------------------------
// Init: zero the h double buffer and the barrier counter (runs every launch,
// keeps graph replays deterministic).
// -------------------------------------------------------------------------
__global__ void k_zero(_Float16* __restrict__ hg, int* __restrict__ bar) {
  int i = blockIdx.x * 512 + threadIdx.x;       // 32768 threads
  ((float*)hg)[i] = 0.0f;                        // 32768 dwords = 128 KB, whole hg region
  if (i == 0) *bar = 0;
}

// -------------------------------------------------------------------------
// Phase 0: swizzle w_hh (f32 [1024][256]) into f16 WMMA B-fragment layout.
//   lane L of frag (nt,ki) holds halfs e=0..15: g = nt*16 + L%16,
//   k = ki*32 + (L/16)*16 + e.   Flat: wb[((nt*8+ki)*32 + L)*16 + e]
// -------------------------------------------------------------------------
__global__ void k_prep(const float* __restrict__ whh, _Float16* __restrict__ wb) {
  int f  = blockIdx.x * 256 + threadIdx.x;   // 0..16383
  int L  = f & 31;
  int ki = (f >> 5) & 7;
  int nt = f >> 8;
  int g  = nt * 16 + (L & 15);
  int kb = ki * 32 + (L >> 4) * 16;
  const float* src = whh + (size_t)g * HH + kb;
  v4f f0 = *(const v4f*)(src);
  v4f f1 = *(const v4f*)(src + 4);
  v4f f2 = *(const v4f*)(src + 8);
  v4f f3 = *(const v4f*)(src + 12);
  v8h lo = {(_Float16)f0[0], (_Float16)f0[1], (_Float16)f0[2], (_Float16)f0[3],
            (_Float16)f1[0], (_Float16)f1[1], (_Float16)f1[2], (_Float16)f1[3]};
  v8h hi = {(_Float16)f2[0], (_Float16)f2[1], (_Float16)f2[2], (_Float16)f2[3],
            (_Float16)f3[0], (_Float16)f3[1], (_Float16)f3[2], (_Float16)f3[3]};
  _Float16* dst = wb + (size_t)f * 16;
  *(v8h*)dst       = lo;
  *(v8h*)(dst + 8) = hi;
}

// -------------------------------------------------------------------------
// Phase 1: pre[m][g] = sum_d x[m][d]*w_ih[g][d] + (b_ih[g]+b_hh[g])
// M=16384, N=1024, K=4096.  f32 -> f16 inline, f32 accum via WMMA.
// Block: 256 thr = 8 waves, tile 128(M) x 64(N); K staged 64 per barrier.
// -------------------------------------------------------------------------
#define AST 72   // LDS row stride in halfs (144B; r*36 dwords mod 64 all distinct)

__global__ __launch_bounds__(256)
void k_igemm(const float* __restrict__ x, const float* __restrict__ w_ih,
             const float* __restrict__ b_ih, const float* __restrict__ b_hh,
             float* __restrict__ pre)
{
  __shared__ alignas(16) _Float16 As[128 * AST];   // 18432 B
  __shared__ alignas(16) _Float16 Bs[ 64 * AST];   //  9216 B

  const int tid   = threadIdx.x;
  const int lane  = tid & 31;
  const int wave  = tid >> 5;
  const int wm    = wave & 3;     // 0..3 : 32-row band
  const int wn    = wave >> 2;    // 0..1 : 32-col band
  const int m0    = blockIdx.x * 128;
  const int n0    = blockIdx.y * 64;
  const int row16 = lane & 15;
  const int half  = lane >> 4;    // 0/1

  v8f acc[2][2] = {};

  for (int k0 = 0; k0 < DD; k0 += 64) {
    #pragma unroll
    for (int i = 0; i < 8; ++i) {
      int idx = i * 256 + tid;
      int r = idx >> 4, c4 = (idx & 15) * 4;
      float4 v = *(const float4*)(x + (size_t)(m0 + r) * DD + k0 + c4);
      v4h h4 = {(_Float16)v.x, (_Float16)v.y, (_Float16)v.z, (_Float16)v.w};
      *(v4h*)(As + r * AST + c4) = h4;
    }
    #pragma unroll
    for (int i = 0; i < 4; ++i) {
      int idx = i * 256 + tid;
      int r = idx >> 4, c4 = (idx & 15) * 4;
      float4 v = *(const float4*)(w_ih + (size_t)(n0 + r) * DD + k0 + c4);
      v4h h4 = {(_Float16)v.x, (_Float16)v.y, (_Float16)v.z, (_Float16)v.w};
      *(v4h*)(Bs + r * AST + c4) = h4;
    }
    if (k0 + 64 < DD) {
      __builtin_prefetch(x + (size_t)(m0 + (tid >> 1)) * DD + k0 + 64 + (tid & 1) * 32, 0, 0);
      if (tid < 128)
        __builtin_prefetch(w_ih + (size_t)(n0 + (tid >> 1)) * DD + k0 + 64 + (tid & 1) * 32, 0, 0);
    }
    __syncthreads();

    v16h a[2][2], b[2][2];
    #pragma unroll
    for (int s = 0; s < 2; ++s) {
      #pragma unroll
      for (int tm = 0; tm < 2; ++tm) {
        const _Float16* p = As + (wm * 32 + tm * 16 + row16) * AST + s * 32 + half * 16;
        a[s][tm] = cat16(*(const v8h*)p, *(const v8h*)(p + 8));
      }
      #pragma unroll
      for (int tn = 0; tn < 2; ++tn) {
        const _Float16* p = Bs + (wn * 32 + tn * 16 + row16) * AST + s * 32 + half * 16;
        b[s][tn] = cat16(*(const v8h*)p, *(const v8h*)(p + 8));
      }
    }
    #pragma unroll
    for (int s = 0; s < 2; ++s)
      #pragma unroll
      for (int tm = 0; tm < 2; ++tm)
        #pragma unroll
        for (int tn = 0; tn < 2; ++tn)
          acc[tm][tn] = __builtin_amdgcn_wmma_f32_16x16x32_f16(
              false, a[s][tm], false, b[s][tn], (short)0, acc[tm][tn], false, false);
    __syncthreads();
  }

  #pragma unroll
  for (int tn = 0; tn < 2; ++tn) {
    int gn = n0 + wn * 32 + tn * 16 + row16;
    float bias = b_ih[gn] + b_hh[gn];
    #pragma unroll
    for (int tm = 0; tm < 2; ++tm) {
      int mb = wm * 32 + tm * 16 + half * 8;
      #pragma unroll
      for (int e = 0; e < 8; ++e) {
        int gm = m0 + mb + e;               // gm = b*256 + t  (x row order)
        pre[(size_t)gm * GG + gn] = acc[tm][tn][e] + bias;
      }
    }
  }
}

// -------------------------------------------------------------------------
extern "C" void kernel_launch(void* const* d_in, const int* in_sizes, int n_in,
                              void* d_out, int out_size, void* d_ws, size_t ws_size,
                              hipStream_t stream)
{
  (void)in_sizes; (void)n_in; (void)out_size; (void)ws_size;
  const float* x    = (const float*)d_in[0];
  const float* w_ih = (const float*)d_in[1];
  const float* w_hh = (const float*)d_in[2];
  const float* b_ih = (const float*)d_in[3];
  const float* b_hh = (const float*)d_in[4];
  const float* w_fc = (const float*)d_in[5];
  const float* b_fc = (const float*)d_in[6];
  float* out = (float*)d_out;

  // workspace layout
  char* ws = (char*)d_ws;
  float*    pre  = (float*)ws;                                   // 64 MB
  _Float16* wb   = (_Float16*)(ws + (size_t)MM * GG * 4);        // 512 KB
  _Float16* hg   = (_Float16*)(ws + (size_t)MM * GG * 4 + 524288);   // 128 KB
  float*    hfin = (float*)   (ws + (size_t)MM * GG * 4 + 524288 + 131072); // 64 KB
  int*      bar  = (int*)     (ws + (size_t)MM * GG * 4 + 524288 + 131072 + 65536);

  k_zero<<<64, 512, 0, stream>>>(hg, bar);
  k_prep<<<64, 256, 0, stream>>>(w_hh, wb);

  dim3 g1(MM / 128, GG / 64);          // 128 x 16 blocks
  k_igemm<<<g1, 256, 0, stream>>>(x, w_ih, b_ih, b_hh, pre);

  k_lstm<<<NWG, 512, SM2, stream>>>(pre, wb, w_fc, b_fc, hg, hfin, bar, out);
}
